// PAM_2654289789670
// MI455X (gfx1250) — matrix operand order repaired and checked by reading, not compiled, not run on hardware
//
#include <hip/hip_runtime.h>

typedef __bf16 bf16_t;
typedef __attribute__((ext_vector_type(16))) __bf16 v16bf;
typedef __attribute__((ext_vector_type(8)))  __bf16 v8bf;
typedef __attribute__((ext_vector_type(8)))  float  v8f;

// ---------------------------------------------------------------------------
// WMMA helpers (CDNA5 wave32, V_WMMA_F32_16X16X32_BF16)
// A (16x32, row-major in LDS): lane l(0..15) -> row M=l; low half-wave holds
//   K={0..7,16..23}, high half K={8..15,24..31} as two contiguous 8-elem runs.
// B^T (rows = N cols of B, 32 K contiguous): lane l -> column N=l; low half
//   K={0..7,16..23}, high half K={8..15,24..31} (same as A).
// C/D: VGPR r, lanes 0-15 -> (M=r, N=lane); lanes 16-31 -> (M=8+r, N=lane-16).
// ---------------------------------------------------------------------------
static __device__ inline v16bf cat8(v8bf a, v8bf b) {
  v16bf r;
#pragma unroll
  for (int i = 0; i < 8; i++) { r[i] = a[i]; r[i + 8] = b[i]; }
  return r;
}

static __device__ inline v16bf load_a16x32(const bf16_t* base, int stride) {
  int lane = threadIdx.x & 31;
  int l = lane & 15, hf = lane >> 4;
  const bf16_t* p = base + l * stride + 8 * hf;
  v8bf lo = *(const v8bf*)p;
  v8bf hi = *(const v8bf*)(p + 16);
  return cat8(lo, hi);
}

static __device__ inline v16bf load_bt32x16(const bf16_t* base, int stride) {
  int lane = threadIdx.x & 31;
  int l = lane & 15, hf = lane >> 4;
  const bf16_t* p = base + l * stride + 8 * hf;
  v8bf lo = *(const v8bf*)p;
  v8bf hi = *(const v8bf*)(p + 16);
  return cat8(lo, hi);
}

static __device__ inline v8f wmma_bf16(v16bf a, v16bf b, v8f c) {
  return __builtin_amdgcn_wmma_f32_16x16x32_bf16(false, a, false, b,
                                                 (short)0, c, false, false);
}

// ---------------------------------------------------------------------------
// Async global -> LDS B128 copy (gfx1250 GLOBAL_LOAD_ASYNC_TO_LDS_B128).
// Per lane: copies 16 bytes from global vaddr to LDS address vdst; tracked by ASYNCcnt.
// LDS byte address = low 32 bits of the flat shared pointer (ISA 10.2).
// ---------------------------------------------------------------------------
static __device__ inline void async_copy_b128(void* lds, const void* gptr) {
  unsigned loff = (unsigned)(size_t)lds;
  asm volatile("global_load_async_to_lds_b128 %0, %1, off"
               :: "v"(loff), "v"(gptr) : "memory");
}
static __device__ inline void async_wait0() {
  asm volatile("s_wait_asynccnt 0x0" ::: "memory");
}

// monotonic int key for float atomicMax
static __device__ inline unsigned fkey(float f) {
  unsigned u = __float_as_uint(f);
  return (u & 0x80000000u) ? ~u : (u | 0x80000000u);
}
static __device__ inline float funkey(unsigned k) {
  unsigned u = (k & 0x80000000u) ? (k & 0x7fffffffu) : ~k;
  return __uint_as_float(u);
}

// ---------------------------------------------------------------------------
// Prep kernels: fp32 -> bf16 conversions / weight re-layout
// ---------------------------------------------------------------------------
__global__ void __launch_bounds__(256) prep_x(const float* __restrict__ xl,
                                              const float* __restrict__ xr,
                                              bf16_t* __restrict__ xb) {
  size_t i = (size_t)blockIdx.x * 256 + threadIdx.x;
  const size_t half = 20971520;  // 8*64*128*320
  if (i < 2 * half) {
    float v = (i < half) ? xl[i] : xr[i - half];
    xb[i] = (bf16_t)v;
  }
}

// rb_w (O=64,I=64,3,3) -> wt[o][tap*64+ci], tap = ky*3+kx (bf16)
__global__ void __launch_bounds__(256) prep_w3(const float* __restrict__ w1,
                                               const float* __restrict__ w2,
                                               bf16_t* __restrict__ wt1,
                                               bf16_t* __restrict__ wt2) {
  int i = blockIdx.x * 256 + threadIdx.x;
  if (i >= 2 * 64 * 576) return;
  int sel = i / 36864;
  int r = i - sel * 36864;
  int o = r / 576, k = r - o * 576;
  int tap = k >> 6, ci = k & 63;
  const float* w = sel ? w2 : w1;
  bf16_t* d = sel ? wt2 : wt1;
  d[o * 576 + k] = (bf16_t)w[((size_t)o * 64 + ci) * 9 + tap];
}

// 1x1 weights (O,I,1,1) -> bf16 (o,ci)
__global__ void __launch_bounds__(256) prep_w1(const float* __restrict__ wl,
                                               const float* __restrict__ wr,
                                               bf16_t* __restrict__ wlb,
                                               bf16_t* __restrict__ wrb) {
  int i = blockIdx.x * 256 + threadIdx.x;
  if (i >= 2 * 4096) return;
  int sel = i >> 12, r = i & 4095;
  const float* w = sel ? wr : wl;
  bf16_t* d = sel ? wrb : wlb;
  d[r] = (bf16_t)w[r];
}

// ---------------------------------------------------------------------------
// Implicit-GEMM conv kernel. One workgroup: 64 x-positions (one row y of one
// image n) x 64 output channels.  KH=3: 3x3 pad 1 (resblock); KH=1: 1x1 proj.
// MODE 0: LeakyReLU(0.1), bf16 NCHW out   (conv1)
// MODE 1: + fp32 residual,  bf16 NCHW out (conv2)
// MODE 2: + bias,           bf16 NHWC out (1x1 projection -> F0/F1)
// ---------------------------------------------------------------------------
template <int KH, int MODE>
__global__ void __launch_bounds__(256) conv_kernel(
    const bf16_t* __restrict__ in, const bf16_t* __restrict__ wt,
    const float* __restrict__ resid_l, const float* __restrict__ resid_r,
    const float* __restrict__ bias, bf16_t* __restrict__ out) {
  constexpr int PAD = KH / 2;
  constexpr int TAPS = KH * KH;
  constexpr int XE = 64 + 2 * PAD;       // staged x extent
  constexpr int KTOT = TAPS * 64;        // GEMM K
  constexpr int WSTR = KTOT + 8;         // padded weight row stride (elems)
  constexpr int SSTR = 72;               // padded im2row stride (elems)

  extern __shared__ char smem[];
  bf16_t* wlds = (bf16_t*)smem;                       // 64*WSTR (B^T rows)
  bf16_t* sin  = wlds + 64 * WSTR;                    // KH*XE*SSTR
  float*  ttile = (float*)(sin + (size_t)KH * XE * SSTR);  // 8 waves * 256

  const int tid = threadIdx.x;
  const int blk = blockIdx.x;
  const int xb5 = blk % 5;
  const int y = (blk / 5) % 128;
  const int n = blk / 640;
  const int x0 = xb5 * 64;

  {  // stage weights: async B128 copies (rows 16B-aligned on both sides)
    constexpr int CPR = KTOT / 8;  // 16B chunks per row
    for (int i = tid; i < 64 * CPR; i += 256) {
      int o = i / CPR, d = i - o * CPR;
      async_copy_b128(wlds + o * WSTR + d * 8, wt + o * KTOT + d * 8);
    }
  }
  {  // stage im2row input: sin[(ky*XE + x')*SSTR + ci]  (halo + zero fill)
    const int tot = KH * 64 * XE;
    for (int i = tid; i < tot; i += 256) {
      int ky = i / (64 * XE);
      int rem = i - ky * 64 * XE;
      int ci = rem / XE, xp = rem - ci * XE;
      int yi = y + ky - PAD;
      int xi = x0 + xp - PAD;
      bf16_t v = (bf16_t)0.0f;
      if ((unsigned)yi < 128u && (unsigned)xi < 320u)
        v = in[(((size_t)n * 64 + ci) * 128 + yi) * 320 + xi];
      sin[(ky * XE + xp) * SSTR + ci] = v;
    }
  }
  async_wait0();
  __syncthreads();

  const int wave = tid >> 5;
  const int lane = tid & 31;
  const int l = lane & 15, hf = lane >> 4;
  const int mt = wave >> 1;            // m-tile 0..3
  const int nt0 = (wave & 1) * 2;      // n-tiles {nt0, nt0+1}

  v8f acc0 = {0, 0, 0, 0, 0, 0, 0, 0};
  v8f acc1 = {0, 0, 0, 0, 0, 0, 0, 0};
#pragma unroll
  for (int kc = 0; kc < KTOT / 32; kc++) {
    const int tap = kc >> 1;
    const int ci0 = (kc & 1) * 32;
    const int ky = (KH == 3) ? tap / 3 : 0;
    const int kx = (KH == 3) ? tap % 3 : 0;
    v16bf a = load_a16x32(sin + ((ky * XE + mt * 16 + kx) * SSTR + ci0), SSTR);
    v16bf b0 = load_bt32x16(wlds + (nt0 * 16) * WSTR + kc * 32, WSTR);
    v16bf b1 = load_bt32x16(wlds + ((nt0 + 1) * 16) * WSTR + kc * 32, WSTR);
    acc0 = wmma_bf16(a, b0, acc0);
    acc1 = wmma_bf16(a, b1, acc1);
  }

  float* tt = ttile + wave * 256;
#pragma unroll
  for (int tno = 0; tno < 2; tno++) {
    v8f acc = tno ? acc1 : acc0;
    const int nt = nt0 + tno;
    if constexpr (MODE == 2) {
      // direct NHWC store: lane-contiguous in channel
#pragma unroll
      for (int r = 0; r < 8; r++) {
        int m = r + 8 * hf;
        int co = nt * 16 + l;
        float v = acc[r] + bias[co];
        int x = x0 + mt * 16 + m;
        out[(((size_t)n * 128 + y) * 320 + x) * 64 + co] = (bf16_t)v;
      }
    } else {
      // transpose through LDS so global stores are x-contiguous
#pragma unroll
      for (int r = 0; r < 8; r++) {
        float v = acc[r];
        if constexpr (MODE == 0) v = v > 0.f ? v : 0.1f * v;
        tt[(r + 8 * hf) * 16 + l] = v;
      }
#pragma unroll
      for (int g = 0; g < 8; g++) {
        int cl = 2 * g + hf;
        int m = l;
        float v = tt[m * 16 + cl];
        int co = nt * 16 + cl;
        int x = x0 + mt * 16 + m;
        size_t idx = (((size_t)n * 64 + co) * 128 + y) * 320 + x;
        if constexpr (MODE == 1) {
          const float* rb = (n < 8) ? resid_l : resid_r;
          v += rb[(((size_t)(n & 7) * 64 + co) * 128 + y) * 320 + x];
        }
        out[idx] = (bf16_t)v;
      }
    }
  }
  (void)resid_l; (void)resid_r; (void)bias;
}

// ---------------------------------------------------------------------------
// Attention kernel: one workgroup per (b,h).  F0/F1/xr rows live in LDS; S is
// recomputed per pass (max pass, sum pass, emit pass) -- never hits HBM.
// ---------------------------------------------------------------------------
__global__ void __launch_bounds__(256) attn_kernel(
    const bf16_t* __restrict__ F0, const bf16_t* __restrict__ F1,
    const bf16_t* __restrict__ xb, float* __restrict__ out_f,
    float* __restrict__ out_V, float* __restrict__ out_Ml,
    float* __restrict__ out_Mr) {
  extern __shared__ char smem[];
  bf16_t* f0s = (bf16_t*)smem;           // 320 x 72 (row-major w,c)  A layout
  bf16_t* f1s = f0s + 320 * 72;          // 320 x 72 (row-major v,c)  B^T layout
  bf16_t* xs  = f1s + 320 * 72;          // 64 x 328 (c,v)            B^T layout
  float* rmf  = (float*)(xs + 64 * 328); // 320 row max (keys then floats)
  float* cmf  = rmf + 320;               // 320 col max
  float* rs   = cmf + 320;               // 320 row expsum -> reciprocal
  float* cs   = rs + 320;                // 320 col expsum -> reciprocal
  float* vacc = cs + 320;                // 320 V accumulator
  bf16_t* msbase = (bf16_t*)(vacc + 320);  // 8 waves * 640 (16x40 A-stage)

  const int tid = threadIdx.x;
  const int bh = blockIdx.x;
  const int bi = bh / 128;
  const int y = bh - bi * 128;
  const int wave = tid >> 5;
  const int lane = tid & 31;
  const int l = lane & 15, hf = lane >> 4;

  // stage F0/F1 rows: async B128 (global rows 128B, LDS stride 144B)
  for (int i = tid; i < 320 * 8; i += 256) {
    int w = i >> 3;
    int c8 = (i & 7) * 8;
    async_copy_b128(f0s + w * 72 + c8, F0 + ((size_t)bh * 320 + w) * 64 + c8);
    async_copy_b128(f1s + w * 72 + c8, F1 + ((size_t)bh * 320 + w) * 64 + c8);
  }
  // stage xr row (right image bi, layout [c][v]): async B128
  for (int i = tid; i < 64 * 40; i += 256) {
    int c = i / 40;
    int o = (i - c * 40) * 8;
    async_copy_b128(xs + c * 328 + o,
                    xb + (((size_t)(8 + bi) * 64 + c) * 128 + y) * 320 + o);
  }
  unsigned* urm = (unsigned*)rmf;
  unsigned* ucm = (unsigned*)cmf;
  for (int i = tid; i < 320; i += 256) {
    urm[i] = 0u; ucm[i] = 0u;
    rs[i] = 0.f; cs[i] = 0.f; vacc[i] = 0.f;
  }
  async_wait0();
  __syncthreads();

  // ---- pass 1a: row/col maxima -------------------------------------------
  for (int t = wave; t < 400; t += 8) {
    int j0 = (t / 20) * 16, i0 = (t % 20) * 16;
    v8f s = {0, 0, 0, 0, 0, 0, 0, 0};
    s = wmma_bf16(load_a16x32(f0s + j0 * 72, 72),
                  load_bt32x16(f1s + i0 * 72, 72), s);
    s = wmma_bf16(load_a16x32(f0s + j0 * 72 + 32, 72),
                  load_bt32x16(f1s + i0 * 72 + 32, 72), s);
#pragma unroll
    for (int r = 0; r < 8; r++) {
      float v = s[r];
      atomicMax(&urm[j0 + r + 8 * hf], fkey(v));
      atomicMax(&ucm[i0 + l], fkey(v));
    }
  }
  __syncthreads();
  for (int i = tid; i < 320; i += 256) {
    float a = funkey(urm[i]);
    float b = funkey(ucm[i]);
    rmf[i] = a; cmf[i] = b;
  }
  __syncthreads();

  // ---- pass 1b: exp sums --------------------------------------------------
  for (int t = wave; t < 400; t += 8) {
    int j0 = (t / 20) * 16, i0 = (t % 20) * 16;
    v8f s = {0, 0, 0, 0, 0, 0, 0, 0};
    s = wmma_bf16(load_a16x32(f0s + j0 * 72, 72),
                  load_bt32x16(f1s + i0 * 72, 72), s);
    s = wmma_bf16(load_a16x32(f0s + j0 * 72 + 32, 72),
                  load_bt32x16(f1s + i0 * 72 + 32, 72), s);
#pragma unroll
    for (int r = 0; r < 8; r++) {
      float v = s[r];
      int j = j0 + r + 8 * hf;
      int i = i0 + l;
      atomicAdd(&rs[j], __expf(v - rmf[j]));
      atomicAdd(&cs[i], __expf(v - cmf[i]));
    }
  }
  __syncthreads();
  for (int i = tid; i < 320; i += 256) {
    rs[i] = 1.f / rs[i];
    cs[i] = 1.f / cs[i];
  }
  __syncthreads();

  // ---- pass 2: emit M_rtl / M_ltr / V, accumulate out = M_rtl @ xr -------
  bf16_t* ms = msbase + wave * 640;  // 16x40 bf16 per-wave A stage
  float* tt = (float*)ms;            // reused as 16x16 f32 transpose buffer
  for (int jb = wave; jb < 20; jb += 8) {
    const int j0 = jb * 16;
    v8f oacc[4];
#pragma unroll
    for (int q = 0; q < 4; q++) oacc[q] = (v8f){0, 0, 0, 0, 0, 0, 0, 0};

    for (int ic = 0; ic < 10; ic++) {
      const int ibase = ic * 32;
#pragma unroll
      for (int sub = 0; sub < 2; sub++) {
        const int i0 = ibase + sub * 16;
        v8f s = {0, 0, 0, 0, 0, 0, 0, 0};
        s = wmma_bf16(load_a16x32(f0s + j0 * 72, 72),
                      load_bt32x16(f1s + i0 * 72, 72), s);
        s = wmma_bf16(load_a16x32(f0s + j0 * 72 + 32, 72),
                      load_bt32x16(f1s + i0 * 72 + 32, 72), s);
#pragma unroll
        for (int r = 0; r < 8; r++) {
          int m = r + 8 * hf;
          int j = j0 + m;
          int i = i0 + l;
          float v = s[r];
          float er = __expf(v - rmf[j]) * rs[j];  // M_rtl[j][i]
          out_Mr[((size_t)bh * 320 + j) * 320 + i] = er;
          float ec = __expf(v - cmf[i]) * cs[i];  // M_ltr[i][j]
          out_Ml[((size_t)bh * 320 + i) * 320 + j] = ec;
          atomicAdd(&vacc[j], ec);
          ms[m * 40 + sub * 16 + l] = (bf16_t)er;
        }
      }
      // out[w][c] += M_rtl[w, ibase..ibase+31] @ xr[ibase..ibase+31, c]
      v16bf am = load_a16x32(ms, 40);
#pragma unroll
      for (int q = 0; q < 4; q++)
        oacc[q] = wmma_bf16(am, load_bt32x16(xs + (q * 16) * 328 + ibase, 328),
                            oacc[q]);
    }
    // write out_f tiles (transpose via LDS -> w-contiguous stores)
#pragma unroll
    for (int q = 0; q < 4; q++) {
#pragma unroll
      for (int r = 0; r < 8; r++) tt[(r + 8 * hf) * 16 + l] = oacc[q][r];
#pragma unroll
      for (int g = 0; g < 8; g++) {
        int cl = 2 * g + hf;
        int m = l;
        float v = tt[m * 16 + cl];
        int c = q * 16 + cl;
        out_f[(((size_t)bi * 64 + c) * 128 + y) * 320 + j0 + m] = v;
      }
    }
  }
  __syncthreads();
  for (int i = tid; i < 320; i += 256)
    out_V[(size_t)bh * 320 + i] = (vacc[i] > 0.1f) ? 1.f : 0.f;
}

// ---------------------------------------------------------------------------
extern "C" void kernel_launch(void* const* d_in, const int* in_sizes, int n_in,
                              void* d_out, int out_size, void* d_ws,
                              size_t ws_size, hipStream_t stream) {
  const float* xl = (const float*)d_in[0];
  const float* xr = (const float*)d_in[1];
  const float* w1 = (const float*)d_in[2];
  const float* w2 = (const float*)d_in[3];
  const float* wl = (const float*)d_in[4];
  const float* bl = (const float*)d_in[5];
  const float* wr = (const float*)d_in[6];
  const float* br = (const float*)d_in[7];

  char* ws = (char*)d_ws;
  bf16_t* xb   = (bf16_t*)ws;                    // 16 imgs bf16: 83,886,080 B
  bf16_t* y1   = (bf16_t*)(ws + 83886080ull);    // conv1 out / later F0,F1
  bf16_t* bufb = (bf16_t*)(ws + 167772160ull);   // resblock out
  bf16_t* wt1  = (bf16_t*)(ws + 251658240ull);   // 64*576
  bf16_t* wt2  = wt1 + 64 * 576;
  bf16_t* wlb  = wt2 + 64 * 576;                 // 64*64
  bf16_t* wrb  = wlb + 64 * 64;
  bf16_t* F0 = y1;                               // y1 dead after conv2
  bf16_t* F1 = y1 + 20971520ull;

  float* out_f = (float*)d_out;
  float* out_V = out_f + 20971520ull;
  float* out_Ml = out_V + 327680ull;
  float* out_Mr = out_Ml + 104857600ull;

  prep_x<<<163840, 256, 0, stream>>>(xl, xr, xb);
  prep_w3<<<288, 256, 0, stream>>>(w1, w2, wt1, wt2);
  prep_w1<<<32, 256, 0, stream>>>(wl, wr, wlb, wrb);

  const size_t lds3 = (size_t)64 * 584 * 2 + (size_t)3 * 66 * 72 * 2 + 8192;
  const size_t lds1 = (size_t)64 * 72 * 2 + (size_t)64 * 72 * 2 + 8192;

  // resblock: conv1 (leaky) then conv2 (+residual) on all 16 images
  conv_kernel<3, 0><<<16 * 128 * 5, 256, lds3, stream>>>(
      xb, wt1, nullptr, nullptr, nullptr, y1);
  conv_kernel<3, 1><<<16 * 128 * 5, 256, lds3, stream>>>(
      y1, wt2, xl, xr, nullptr, bufb);

  // 1x1 projections -> F0 (left, wl/bl) and F1 (right, wr/br), NHWC bf16
  conv_kernel<1, 2><<<8 * 128 * 5, 256, lds1, stream>>>(
      bufb, wlb, nullptr, nullptr, bl, F0);
  conv_kernel<1, 2><<<8 * 128 * 5, 256, lds1, stream>>>(
      bufb + 20971520ull, wrb, nullptr, nullptr, br, F1);

  // per-(b,h) attention
  const size_t ldsA = (size_t)320 * 72 * 2 * 2 + (size_t)64 * 328 * 2 +
                      5 * 320 * 4 + 8 * 640 * 2;
  attn_kernel<<<1024, 256, ldsA, stream>>>(F0, F1, xb, out_f, out_V, out_Ml,
                                           out_Mr);
  (void)in_sizes; (void)n_in; (void)out_size; (void)ws_size;
}
